// VariableSelection_3564822856103
// MI455X (gfx1250) — matrix operand order repaired and checked by reading, not compiled, not run on hardware
//
#include <hip/hip_runtime.h>
#include <hip/hip_bf16.h>
#include <math.h>

// ---------------------------------------------------------------------------
// VariableSelection (TFT) on MI455X / gfx1250.
// B=128 T=1024 N=10 D=40, fp32 in/out.
// All GEMMs run on v_wmma_f32_16x16x32_f16 (f16 operands, f32 accumulate).
// All A-operand chunks are compile-time fully-valid or fully-zero: no
// per-element predication (K dims are multiples of 8; staging buffers are
// stride-64 with permanently-zero pad columns 40..63).
// ---------------------------------------------------------------------------

#define BB   128
#define TT   1024
#define NN   10
#define DD   40
#define NDK  400          // N*D
#define BT   (BB*TT)      // 131072 tokens
#define TN   (TT*NN)      // 10240

typedef __attribute__((ext_vector_type(16))) _Float16 v16h;
typedef __attribute__((ext_vector_type(8)))  float    v8f;

static __device__ __forceinline__ v8f vzero8() {
    v8f z;
#pragma unroll
    for (int i = 0; i < 8; ++i) z[i] = 0.f;
    return z;
}

static __device__ __forceinline__ float eluf(float x) {
    return x > 0.f ? x : (expf(x) - 1.f);
}
static __device__ __forceinline__ float sigmf(float x) {
    return 1.f / (1.f + expf(-x));
}

// Drain LDS ops; wave32 lockstep + in-order DS => cross-lane LDS handoff safe.
static __device__ __forceinline__ void wave_sync() {
    asm volatile("s_wait_dscnt 0" ::: "memory");
}

// D = A*B + C, 16x16x32 f16->f32
static __device__ __forceinline__ v8f wmma16(v16h a, v16h b, v8f c) {
    return __builtin_amdgcn_wmma_f32_16x16x32_f16(
        /*neg_a=*/false, a, /*neg_b=*/false, b,
        /*c_mod=*/(short)0, c, /*reuse_a=*/false, /*reuse_b=*/false);
}

static __device__ __forceinline__ void cvt8(v16h& A, int base, float4 x,
                                            float4 y) {
    A[base + 0] = (_Float16)x.x;
    A[base + 1] = (_Float16)x.y;
    A[base + 2] = (_Float16)x.z;
    A[base + 3] = (_Float16)x.w;
    A[base + 4] = (_Float16)y.x;
    A[base + 5] = (_Float16)y.y;
    A[base + 6] = (_Float16)y.z;
    A[base + 7] = (_Float16)y.w;
}

// A operand (16x32 f16), fully valid: rp = this lane's row base (f32, global
// or LDS, 16B aligned). Lane l<16: row l, K={kb..kb+7, kb+16..23}; lane l+16:
// K={kb+8..15, kb+24..31}.
static __device__ __forceinline__ v16h loadA_full(const float* rp, int kb,
                                                  int hi) {
    const float* p0 = rp + kb + hi * 8;
    const float* p1 = rp + kb + 16 + hi * 8;
    v16h A;
    cvt8(A, 0, *(const float4*)p0, *(const float4*)(p0 + 4));
    cvt8(A, 8, *(const float4*)p1, *(const float4*)(p1 + 4));
    return A;
}

// A operand where only the first 16 K's of the 32-tile are valid (K1 tail:
// kb=384, K=400). Chunk0 valid for every lane, chunk1 zero for every lane.
static __device__ __forceinline__ v16h loadA_tail16(const float* rp, int kb,
                                                    int hi) {
    const float* p0 = rp + kb + hi * 8;
    v16h A;
    cvt8(A, 0, *(const float4*)p0, *(const float4*)(p0 + 4));
#pragma unroll
    for (int i = 8; i < 16; ++i) A[i] = (_Float16)0.f;
    return A;
}

// B operand (32x16 f16) from LDS weight stored transposed: WT[col][Kpad].
// Lane l: col = nbase + (l&15); halfs 0..7 = K { kb + (l>>4)*8 + 0..7 },
// halfs 8..15 = K { kb + 16 + (l>>4)*8 + 0..7 }.
static __device__ __forceinline__ v16h loadB_lds(const _Float16* WT, int Kpad,
                                                 int nbase, int kb, int l15,
                                                 int hi) {
    const _Float16* p = WT + (size_t)(nbase + l15) * Kpad + kb + hi * 8;
    v16h r;
    ((uint4*)&r)[0] = ((const uint4*)p)[0];
    ((uint4*)&r)[1] = ((const uint4*)(p + 16))[0];
    return r;
}

// ---------------------------------------------------------------------------
// Kernel 1: grn_v -> logits[B*T, 10]   (wave = 16 tokens)
// ---------------------------------------------------------------------------
__global__ __launch_bounds__(128) void vsn_grn_v_kernel(
    const float* __restrict__ ksi, const float* __restrict__ cctx,
    const float* __restrict__ W2, const float* __restrict__ b2,
    const float* __restrict__ W3, const float* __restrict__ W1,
    const float* __restrict__ b1, const float* __restrict__ Ws,
    const float* __restrict__ bs, const float* __restrict__ W4,
    const float* __restrict__ b4, const float* __restrict__ W5,
    const float* __restrict__ b5, const float* __restrict__ gma,
    const float* __restrict__ bta, float* __restrict__ logits) {
    // Weights transposed to [col][Kpad] f16 (B-operand friendly).
    alignas(16) __shared__ _Float16 sW2T[48 * 416];
    alignas(16) __shared__ _Float16 sWsT[16 * 416];
    alignas(16) __shared__ _Float16 sW3T[48 * 64];
    alignas(16) __shared__ _Float16 sW1T[48 * 64];
    alignas(16) __shared__ _Float16 sW4T[16 * 64];
    alignas(16) __shared__ _Float16 sW5T[16 * 64];
    alignas(16) __shared__ float sStage[4][16 * 64];  // stride 64, cols 40..63 stay 0
    __shared__ float sY[4][16 * 16];

    const int tid = threadIdx.x;
    for (int i = tid; i < 48 * 416; i += 128) {
        int n = i / 416, k = i % 416;
        sW2T[i] = (n < 40 && k < 400) ? (_Float16)W2[k * 40 + n] : (_Float16)0.f;
    }
    for (int i = tid; i < 16 * 416; i += 128) {
        int n = i / 416, k = i % 416;
        sWsT[i] = (n < 10 && k < 400) ? (_Float16)Ws[k * 10 + n] : (_Float16)0.f;
    }
    for (int i = tid; i < 48 * 64; i += 128) {
        int n = i / 64, k = i % 64;
        sW3T[i] = (n < 40 && k < 40) ? (_Float16)W3[k * 40 + n] : (_Float16)0.f;
        sW1T[i] = (n < 40 && k < 40) ? (_Float16)W1[k * 40 + n] : (_Float16)0.f;
    }
    for (int i = tid; i < 16 * 64; i += 128) {
        int n = i / 64, k = i % 64;
        sW4T[i] = (n < 10 && k < 40) ? (_Float16)W4[k * 10 + n] : (_Float16)0.f;
        sW5T[i] = (n < 10 && k < 40) ? (_Float16)W5[k * 10 + n] : (_Float16)0.f;
    }

    const int wave = tid >> 5, lane = tid & 31, l15 = lane & 15,
              hi = (lane >> 4) & 1;
    float* st = sStage[wave];
    float* yb = sY[wave];
    for (int i = lane; i < 16 * 64; i += 32) st[i] = 0.f;  // pad zero-init
    __syncthreads();

    const int slot = blockIdx.x * 4 + wave;  // 1024 wave slots

    for (int it = 0; it < (BT / 16) / 1024; ++it) {  // 8 iters
        const int tile = it * 1024 + slot;
        const int tok0 = tile * 16;
        const int myTok = tok0 + l15;
        const float* xrow = ksi + (size_t)myTok * NDK;

        v8f a0 = vzero8(), a1 = vzero8(), a2 = vzero8(), ar = vzero8();
        // KSI @ W2 (K=400) and KSI @ Ws, sharing the A tile.
#pragma unroll 4
        for (int kk = 0; kk < 12; ++kk) {  // fully-valid tiles
            const int kb = kk * 32;
            v16h A = loadA_full(xrow, kb, hi);
            a0 = wmma16(A, loadB_lds(sW2T, 416, 0, kb, l15, hi), a0);
            a1 = wmma16(A, loadB_lds(sW2T, 416, 16, kb, l15, hi), a1);
            a2 = wmma16(A, loadB_lds(sW2T, 416, 32, kb, l15, hi), a2);
            ar = wmma16(A, loadB_lds(sWsT, 416, 0, kb, l15, hi), ar);
        }
        {  // tail tile: K 384..399 valid, 400..415 zero (uniform, no predicates)
            const int kb = 384;
            v16h A = loadA_tail16(xrow, kb, hi);
            a0 = wmma16(A, loadB_lds(sW2T, 416, 0, kb, l15, hi), a0);
            a1 = wmma16(A, loadB_lds(sW2T, 416, 16, kb, l15, hi), a1);
            a2 = wmma16(A, loadB_lds(sW2T, 416, 32, kb, l15, hi), a2);
            ar = wmma16(A, loadB_lds(sWsT, 416, 0, kb, l15, hi), ar);
        }
        // stage the c tile (coalesced) then unchecked A loads from LDS
        for (int i = lane; i < 16 * 40; i += 32) {
            int m = i / 40, col = i % 40;
            st[m * 64 + col] = cctx[(size_t)(tok0 + m) * DD + col];
        }
        wave_sync();
#pragma unroll
        for (int kk = 0; kk < 2; ++kk) {
            const int kb = kk * 32;
            v16h A = loadA_full(st + l15 * 64, kb, hi);
            a0 = wmma16(A, loadB_lds(sW3T, 64, 0, kb, l15, hi), a0);
            a1 = wmma16(A, loadB_lds(sW3T, 64, 16, kb, l15, hi), a1);
            a2 = wmma16(A, loadB_lds(sW3T, 64, 32, kb, l15, hi), a2);
        }
        wave_sync();
        // eta2 = elu(. + b2) -> stage
#pragma unroll
        for (int r = 0; r < 8; ++r) {
            const int m = r + 8 * hi;
            st[m * 64 + l15] = eluf(a0[r] + b2[l15]);
            st[m * 64 + 16 + l15] = eluf(a1[r] + b2[16 + l15]);
            const int n2 = 32 + l15;
            st[m * 64 + n2] = (n2 < 40) ? eluf(a2[r] + b2[n2]) : 0.f;
        }
        wave_sync();
        // eta1 = eta2 @ W1 + b1
        v8f e0 = vzero8(), e1 = vzero8(), e2 = vzero8();
#pragma unroll
        for (int kk = 0; kk < 2; ++kk) {
            const int kb = kk * 32;
            v16h A = loadA_full(st + l15 * 64, kb, hi);
            e0 = wmma16(A, loadB_lds(sW1T, 64, 0, kb, l15, hi), e0);
            e1 = wmma16(A, loadB_lds(sW1T, 64, 16, kb, l15, hi), e1);
            e2 = wmma16(A, loadB_lds(sW1T, 64, 32, kb, l15, hi), e2);
        }
        wave_sync();
#pragma unroll
        for (int r = 0; r < 8; ++r) {
            const int m = r + 8 * hi;
            st[m * 64 + l15] = e0[r] + b1[l15];
            st[m * 64 + 16 + l15] = e1[r] + b1[16 + l15];
            const int n2 = 32 + l15;
            st[m * 64 + n2] = (n2 < 40) ? (e2[r] + b1[n2]) : 0.f;
        }
        wave_sync();
        // GLU gates: eta1 @ W4 / W5   (N=10 -> one n-tile)
        v8f p4 = vzero8(), p5 = vzero8();
#pragma unroll
        for (int kk = 0; kk < 2; ++kk) {
            const int kb = kk * 32;
            v16h A = loadA_full(st + l15 * 64, kb, hi);
            p4 = wmma16(A, loadB_lds(sW4T, 64, 0, kb, l15, hi), p4);
            p5 = wmma16(A, loadB_lds(sW5T, 64, 0, kb, l15, hi), p5);
        }
        wave_sync();
        // y = resid + glu
#pragma unroll
        for (int r = 0; r < 8; ++r) {
            const int m = r + 8 * hi;
            const int n = l15;
            if (n < 10) {
                float y = (ar[r] + bs[n]) +
                          sigmf(p4[r] + b4[n]) * (p5[r] + b5[n]);
                yb[m * 16 + n] = y;
            }
        }
        wave_sync();
        // LayerNorm over N=10, one lane per token row
        if (lane < 16) {
            const int m = lane;
            const int token = tok0 + m;
            float mean = 0.f;
#pragma unroll
            for (int n = 0; n < 10; ++n) mean += yb[m * 16 + n];
            mean *= 0.1f;
            float var = 0.f;
#pragma unroll
            for (int n = 0; n < 10; ++n) {
                float d = yb[m * 16 + n] - mean;
                var += d * d;
            }
            var *= 0.1f;
            const float rstd = rsqrtf(var + 1e-5f);
#pragma unroll
            for (int n = 0; n < 10; ++n)
                logits[(size_t)token * NN + n] =
                    (yb[m * 16 + n] - mean) * rstd * gma[n] + bta[n];
        }
        wave_sync();
    }
}

// ---------------------------------------------------------------------------
// Kernel 2: softmax over the BATCH axis (legacy dim=0), in place.
// ---------------------------------------------------------------------------
__global__ void vsn_softmax_b(float* __restrict__ lg) {
    const int tn = blockIdx.x * blockDim.x + threadIdx.x;
    if (tn >= TN) return;
    float mx = -3.402823466e38f;
    for (int b = 0; b < BB; ++b)
        mx = fmaxf(mx, lg[(size_t)b * TN + tn]);
    float s = 0.f;
    for (int b = 0; b < BB; ++b)
        s += expf(lg[(size_t)b * TN + tn] - mx);
    const float inv = 1.f / s;
    for (int b = 0; b < BB; ++b) {
        const size_t idx = (size_t)b * TN + tn;
        lg[idx] = expf(lg[idx] - mx) * inv;
    }
}

// ---------------------------------------------------------------------------
// Kernel 3: shared per-variable GRN + weighted combine.
// Wave = 8 tokens; 5 tiles of (8 tokens x 2 variables) = 16 A rows each.
// ---------------------------------------------------------------------------
__global__ __launch_bounds__(128) void vsn_grn_k_kernel(
    const float* __restrict__ ksi, const float* __restrict__ vsel,
    const float* __restrict__ W2, const float* __restrict__ b2,
    const float* __restrict__ W1, const float* __restrict__ b1,
    const float* __restrict__ W4, const float* __restrict__ b4,
    const float* __restrict__ W5, const float* __restrict__ b5,
    const float* __restrict__ gma, const float* __restrict__ bta,
    float* __restrict__ out) {
    alignas(16) __shared__ _Float16 sW2T[48 * 64];
    alignas(16) __shared__ _Float16 sW1T[48 * 64];
    alignas(16) __shared__ _Float16 sW4T[48 * 64];
    alignas(16) __shared__ _Float16 sW5T[48 * 64];
    alignas(16) __shared__ float sX[4][16 * 64];      // input tile, pad = 0
    alignas(16) __shared__ float sStage[4][16 * 64];  // pad cols 40..63 = 0
    __shared__ float sAcc[4][8 * 40];

    const int tid = threadIdx.x;
    for (int i = tid; i < 48 * 64; i += 128) {
        int n = i / 64, k = i % 64;
        bool ok = (n < 40 && k < 40);
        sW2T[i] = ok ? (_Float16)W2[k * 40 + n] : (_Float16)0.f;
        sW1T[i] = ok ? (_Float16)W1[k * 40 + n] : (_Float16)0.f;
        sW4T[i] = ok ? (_Float16)W4[k * 40 + n] : (_Float16)0.f;
        sW5T[i] = ok ? (_Float16)W5[k * 40 + n] : (_Float16)0.f;
    }

    const int wave = tid >> 5, lane = tid & 31, l15 = lane & 15,
              hi = (lane >> 4) & 1;
    float* st = sStage[wave];
    float* sx = sX[wave];
    float* oa = sAcc[wave];
    for (int i = lane; i < 16 * 64; i += 32) { st[i] = 0.f; sx[i] = 0.f; }
    __syncthreads();

    const int slot = blockIdx.x * 4 + wave;

    for (int it = 0; it < (BT / 8) / 1024; ++it) {  // 16 iters
        const int tile = it * 1024 + slot;
        const int tok8 = tile * 8;
        for (int i = lane; i < 8 * 40; i += 32) oa[i] = 0.f;
        wave_sync();

        for (int nb = 0; nb < 5; ++nb) {  // variable pairs
            // A row m -> (token j = m>>1, variable n = 2*nb + (m&1))
            // Stage the 16 ksi rows (coalesced); reused for residual too.
            for (int i = lane; i < 16 * 40; i += 32) {
                int m = i / 40, col = i % 40;
                size_t ridx = (size_t)(tok8 + (m >> 1)) * NN + nb * 2 + (m & 1);
                sx[m * 64 + col] = ksi[ridx * DD + col];
            }
            wave_sync();

            // eta2 = elu(ksi @ W2 + b2)
            v8f a0 = vzero8(), a1 = vzero8(), a2 = vzero8();
#pragma unroll
            for (int kk = 0; kk < 2; ++kk) {
                const int kb = kk * 32;
                v16h A = loadA_full(sx + l15 * 64, kb, hi);
                a0 = wmma16(A, loadB_lds(sW2T, 64, 0, kb, l15, hi), a0);
                a1 = wmma16(A, loadB_lds(sW2T, 64, 16, kb, l15, hi), a1);
                a2 = wmma16(A, loadB_lds(sW2T, 64, 32, kb, l15, hi), a2);
            }
#pragma unroll
            for (int r = 0; r < 8; ++r) {
                const int m = r + 8 * hi;
                st[m * 64 + l15] = eluf(a0[r] + b2[l15]);
                st[m * 64 + 16 + l15] = eluf(a1[r] + b2[16 + l15]);
                const int n2 = 32 + l15;
                st[m * 64 + n2] = (n2 < 40) ? eluf(a2[r] + b2[n2]) : 0.f;
            }
            wave_sync();
            // eta1 = eta2 @ W1 + b1
            v8f e0 = vzero8(), e1 = vzero8(), e2 = vzero8();
#pragma unroll
            for (int kk = 0; kk < 2; ++kk) {
                const int kb = kk * 32;
                v16h A = loadA_full(st + l15 * 64, kb, hi);
                e0 = wmma16(A, loadB_lds(sW1T, 64, 0, kb, l15, hi), e0);
                e1 = wmma16(A, loadB_lds(sW1T, 64, 16, kb, l15, hi), e1);
                e2 = wmma16(A, loadB_lds(sW1T, 64, 32, kb, l15, hi), e2);
            }
            wave_sync();
#pragma unroll
            for (int r = 0; r < 8; ++r) {
                const int m = r + 8 * hi;
                st[m * 64 + l15] = e0[r] + b1[l15];
                st[m * 64 + 16 + l15] = e1[r] + b1[16 + l15];
                const int n2 = 32 + l15;
                st[m * 64 + n2] = (n2 < 40) ? (e2[r] + b1[n2]) : 0.f;
            }
            wave_sync();
            // GLU gates (40 outputs each)
            v8f p4[3], p5[3];
#pragma unroll
            for (int t = 0; t < 3; ++t) { p4[t] = vzero8(); p5[t] = vzero8(); }
#pragma unroll
            for (int kk = 0; kk < 2; ++kk) {
                const int kb = kk * 32;
                v16h A = loadA_full(st + l15 * 64, kb, hi);
#pragma unroll
                for (int t = 0; t < 3; ++t) {
                    p4[t] = wmma16(A, loadB_lds(sW4T, 64, t * 16, kb, l15, hi), p4[t]);
                    p5[t] = wmma16(A, loadB_lds(sW5T, 64, t * 16, kb, l15, hi), p5[t]);
                }
            }
            wave_sync();
            // tilde_pre = ksi + glu (ksi from the staged tile)
#pragma unroll
            for (int r = 0; r < 8; ++r) {
                const int m = r + 8 * hi;
#pragma unroll
                for (int t = 0; t < 3; ++t) {
                    const int col = l15 + 16 * t;
                    if (col < 40) {
                        float glu = sigmf(p4[t][r] + b4[col]) *
                                    (p5[t][r] + b5[col]);
                        st[m * 64 + col] = sx[m * 64 + col] + glu;
                    }
                }
            }
            wave_sync();
            // LayerNorm over D=40 + scaled accumulate into out tile
            float mean = 0.f, rstd = 0.f, w = 0.f;
            int jj = (lane & 15) >> 1;
            if (lane < 16) {
                const int m = lane;
                const int nn = nb * 2 + (m & 1);
                for (int d = 0; d < 40; ++d) mean += st[m * 64 + d];
                mean *= (1.f / 40.f);
                float var = 0.f;
                for (int d = 0; d < 40; ++d) {
                    float dv = st[m * 64 + d] - mean;
                    var += dv * dv;
                }
                var *= (1.f / 40.f);
                rstd = rsqrtf(var + 1e-5f);
                w = vsel[(size_t)(tok8 + jj) * NN + nn];
            }
#pragma unroll
            for (int p = 0; p < 2; ++p) {  // parity split: no LDS conflicts
                if (lane < 16 && (lane & 1) == p) {
                    const int m = lane;
                    for (int d = 0; d < 40; ++d) {
                        float val =
                            (st[m * 64 + d] - mean) * rstd * gma[d] + bta[d];
                        oa[jj * 40 + d] += w * val;
                    }
                }
                wave_sync();
            }
        }
        // flush out tile
        for (int i = lane; i < 8 * 40; i += 32)
            out[(size_t)(tok8 + i / 40) * DD + (i % 40)] = oa[i];
        wave_sync();
    }
}

// ---------------------------------------------------------------------------
extern "C" void kernel_launch(void* const* d_in, const int* in_sizes, int n_in,
                              void* d_out, int out_size, void* d_ws,
                              size_t ws_size, hipStream_t stream) {
    const float* ksi  = (const float*)d_in[0];
    const float* cctx = (const float*)d_in[1];
    const float* vW2  = (const float*)d_in[2];
    const float* vb2  = (const float*)d_in[3];
    const float* vW3  = (const float*)d_in[4];
    const float* vW1  = (const float*)d_in[5];
    const float* vb1  = (const float*)d_in[6];
    const float* vWs  = (const float*)d_in[7];
    const float* vbs  = (const float*)d_in[8];
    const float* vW4  = (const float*)d_in[9];
    const float* vb4  = (const float*)d_in[10];
    const float* vW5  = (const float*)d_in[11];
    const float* vb5  = (const float*)d_in[12];
    const float* vg   = (const float*)d_in[13];
    const float* vbln = (const float*)d_in[14];
    const float* kW2  = (const float*)d_in[15];
    const float* kb2  = (const float*)d_in[16];
    const float* kW1  = (const float*)d_in[17];
    const float* kb1  = (const float*)d_in[18];
    const float* kW4  = (const float*)d_in[19];
    const float* kb4  = (const float*)d_in[20];
    const float* kW5  = (const float*)d_in[21];
    const float* kb5  = (const float*)d_in[22];
    const float* kg   = (const float*)d_in[23];
    const float* kbln = (const float*)d_in[24];
    float* outp = (float*)d_out;

    float* logits = (float*)d_ws;  // B*T*N floats = 5.24 MB

    vsn_grn_v_kernel<<<256, 128, 0, stream>>>(
        ksi, cctx, vW2, vb2, vW3, vW1, vb1, vWs, vbs, vW4, vb4, vW5, vb5, vg,
        vbln, logits);
    vsn_softmax_b<<<(TN + 255) / 256, 256, 0, stream>>>(logits);
    vsn_grn_k_kernel<<<256, 128, 0, stream>>>(
        ksi, logits, kW2, kb2, kW1, kb1, kW4, kb4, kW5, kb5, kg, kbln, outp);
}
